// TtAttention_10153302688289
// MI455X (gfx1250) — matrix operand order repaired and checked by reading, not compiled, not run on hardware
//
#include <hip/hip_runtime.h>
#include <hip/hip_bf16.h>

// ---------------------------------------------------------------------------
// MI455X (gfx1250) GQA attention block.
//  - bf16 v_wmma_f32_16x16x32_bf16 for all matmuls, f32 accumulation
//  - LDS-staged GEMM (128x256 block tile, 64x64 wave tile = 16 WMMA/k-step)
//    fed by GLOBAL_LOAD_ASYNC_TO_LDS_B128 (ASYNCcnt) when available
//  - flash-attention inner loop (no materialized 512MB score tensor)
// ---------------------------------------------------------------------------

#define S_LEN   2048
#define D_MODEL 4096
#define NH      32
#define NKVH    8
#define HDIM    128
#define REPEATS (NH / NKVH)
#define WIN     2048
#define QDIM    (NH * HDIM)    // 4096
#define KVDIM   (NKVH * HDIM)  // 1024
#define ATT_SCALE 0.08838834764831845f  // 128^-0.5
#define MASK_NEG  -1000000000.0f

typedef __attribute__((ext_vector_type(16))) __bf16          v16bf;
typedef __attribute__((ext_vector_type(8)))  float           v8f;
typedef __attribute__((ext_vector_type(8)))  unsigned short  ushort8;
typedef __attribute__((ext_vector_type(4)))  int             v4i;

struct frag32 { ushort8 lo, hi; };  // 32 bytes == v16bf

#define AS_GLOBAL __attribute__((address_space(1)))
#define AS_LDS    __attribute__((address_space(3)))

#if __has_builtin(__builtin_amdgcn_global_load_async_to_lds_b128)
#define USE_ASYNC_LDS 1
#else
#define USE_ASYNC_LDS 0
#endif

__device__ __forceinline__ unsigned short f32_to_bf16(float f) {
  union { float f; unsigned int u; } v; v.f = f;
  unsigned int u = v.u;
  unsigned int r = (u + 0x7FFFu + ((u >> 16) & 1u)) >> 16;   // RNE
  return (unsigned short)r;
}
__device__ __forceinline__ float bf16_to_f32(unsigned short h) {
  union { unsigned int u; float f; } v;
  v.u = ((unsigned int)h) << 16;
  return v.f;
}

__device__ __forceinline__ v8f wmma_bf16(v16bf a, v16bf b, v8f c) {
  // (neg_a, A, neg_b, B, c_mod, C, reuse_a, reuse_b)
  return __builtin_amdgcn_wmma_f32_16x16x32_bf16(false, a, false, b,
                                                 (short)0, c, false, false);
}

// A-fragment (16x32, 16-bit): lane m=lane%16, kg=lane/16:
//   elems 0..15 = K[k0 + kg*16 .. +15]
__device__ __forceinline__ v16bf load_a_frag(const unsigned short* rowp_k0, int lh) {
  frag32 f;
  f.lo = *(const ushort8*)(rowp_k0 + lh * 16);
  f.hi = *(const ushort8*)(rowp_k0 + lh * 16 + 8);
  return __builtin_bit_cast(v16bf, f);
}
// B-fragment (32x16, 16-bit): lane n=lane%16, half=lane/16:
//   elems 0..15 = K[k0 + half*16 .. +15] of column n  (pass ptr at k0+half*16)
__device__ __forceinline__ v16bf load_b_frag(const unsigned short* colp) {
  frag32 f;
  f.lo = *(const ushort8*)(colp);
  f.hi = *(const ushort8*)(colp + 8);
  return __builtin_bit_cast(v16bf, f);
}

__device__ __forceinline__ v8f v8f_zero() {
  v8f z = {0.f, 0.f, 0.f, 0.f, 0.f, 0.f, 0.f, 0.f};
  return z;
}

// 16-byte global -> LDS copy. Async (ASYNCcnt) when available.
// Pointer materialization via integer casts:
//  - generic global pointer bits == AS1 pointer bits
//  - generic LDS address low 32 bits == LDS byte offset (flat aperture rule)
__device__ __forceinline__ void copy16_g2l(const unsigned short* gsrc,
                                           unsigned short* ldst) {
#if USE_ASYNC_LDS
  __builtin_amdgcn_global_load_async_to_lds_b128(
      (AS_GLOBAL v4i*)(unsigned long long)gsrc,
      (AS_LDS v4i*)(unsigned int)(unsigned long long)ldst,
      0, 0);
#else
  ushort8 tmp = *(const ushort8*)gsrc;
  *(ushort8*)ldst = tmp;
#endif
}

__device__ __forceinline__ void g2l_wait() {
#if USE_ASYNC_LDS
#if __has_builtin(__builtin_amdgcn_s_wait_asynccnt)
  __builtin_amdgcn_s_wait_asynccnt(0);
#else
  asm volatile("s_wait_asynccnt 0" ::: "memory");
#endif
#endif
}

// ---------------------------------------------------------------------------
// f32 -> bf16 cast
// ---------------------------------------------------------------------------
__global__ void cast_f32_bf16_kernel(const float* __restrict__ in,
                                     unsigned short* __restrict__ out, int n) {
  int i = blockIdx.x * blockDim.x + threadIdx.x;
  if (i < n) out[i] = f32_to_bf16(in[i]);
}

// ---------------------------------------------------------------------------
// C[M,N] = A[M,K] * Bt[N,K]^T   (bf16 operands, f32 accumulate)
// Block: 256 threads (8 waves, 2x4), block tile 128x256, wave tile 64x64.
// Per k-step (32): stage A(128x32)=8KB + B(256x32)=16KB into LDS with async
// copies, then each wave does 16 v_wmma from ds_load_b128 fragments.
// ---------------------------------------------------------------------------
__device__ __forceinline__ void store_out(float* C, size_t i, float v)          { C[i] = v; }
__device__ __forceinline__ void store_out(unsigned short* C, size_t i, float v) { C[i] = f32_to_bf16(v); }

template <typename OutT>
__global__ void __launch_bounds__(256)
gemm_bt_wmma(const unsigned short* __restrict__ A,
             const unsigned short* __restrict__ Bt,
             OutT* __restrict__ C,
             int M, int N, int K) {
  __shared__ __align__(16) unsigned short Alds[128 * 32];  // 8 KB
  __shared__ __align__(16) unsigned short Blds[256 * 32];  // 16 KB

  const int tid  = threadIdx.x;
  const int lane = tid & 31;
  const int wave = tid >> 5;
  const int lm = lane & 15;
  const int lh = lane >> 4;
  const int wrow = wave >> 2;              // 0..1  -> 64 rows each
  const int wcol = wave & 3;               // 0..3  -> 64 cols each
  const int mblk = blockIdx.y * 128;
  const int nblk = blockIdx.x * 256;

  v8f acc[4][4];
  #pragma unroll
  for (int mi = 0; mi < 4; ++mi)
    #pragma unroll
    for (int ni = 0; ni < 4; ++ni) acc[mi][ni] = v8f_zero();

  for (int k0 = 0; k0 < K; k0 += 32) {
    // ---- stage tiles (16B chunks; A: 512 chunks, B: 1024 chunks) ----
    #pragma unroll
    for (int i = 0; i < 2; ++i) {
      int c = tid + i * 256;
      int row = c >> 2;
      int col = (c & 3) * 8;
      copy16_g2l(A + (size_t)(mblk + row) * K + k0 + col, Alds + c * 8);
    }
    #pragma unroll
    for (int i = 0; i < 4; ++i) {
      int c = tid + i * 256;
      int row = c >> 2;
      int col = (c & 3) * 8;
      copy16_g2l(Bt + (size_t)(nblk + row) * K + k0 + col, Blds + c * 8);
    }
    g2l_wait();
    __syncthreads();

    // ---- 16 WMMA per wave per k-step ----
    v16bf af[4];
    #pragma unroll
    for (int mi = 0; mi < 4; ++mi)
      af[mi] = load_a_frag(&Alds[(wrow * 64 + mi * 16 + lm) * 32], lh);
    #pragma unroll
    for (int ni = 0; ni < 4; ++ni) {
      v16bf bf = load_b_frag(&Blds[(wcol * 64 + ni * 16 + lm) * 32 + lh * 16]);
      #pragma unroll
      for (int mi = 0; mi < 4; ++mi)
        acc[mi][ni] = wmma_bf16(af[mi], bf, acc[mi][ni]);
    }
    __syncthreads();
  }

  // C/D layout: elem r -> row = r + 8*(lane/16), col = lane%16
  #pragma unroll
  for (int mi = 0; mi < 4; ++mi)
    #pragma unroll
    for (int ni = 0; ni < 4; ++ni)
      #pragma unroll
      for (int r = 0; r < 8; ++r) {
        size_t row = (size_t)(mblk + wrow * 64 + mi * 16 + r + 8 * lh);
        size_t col = (size_t)(nblk + wcol * 64 + ni * 16 + lm);
        store_out(C, row * (size_t)N + col, acc[mi][ni][r]);
      }
}

// ---------------------------------------------------------------------------
// RoPE (interleaved pairs), bf16 in-place, f32 math
// ---------------------------------------------------------------------------
__global__ void rope_kernel(unsigned short* __restrict__ t,
                            const float* __restrict__ cosb,
                            const float* __restrict__ sinb,
                            int nheads, int npairs_total) {
  int idx = blockIdx.x * blockDim.x + threadIdx.x;
  if (idx >= npairs_total) return;
  const int hp = HDIM / 2;                 // 64
  int p = idx % hp;
  int tmp = idx / hp;
  int h = tmp % nheads;
  int s = tmp / nheads;
  size_t base = ((size_t)s * nheads + h) * HDIM + 2 * p;
  float xr = bf16_to_f32(t[base]);
  float xi = bf16_to_f32(t[base + 1]);
  float c = cosb[(size_t)s * hp + p];
  float sn = sinb[(size_t)s * hp + p];
  t[base]     = f32_to_bf16(xr * c - xi * sn);
  t[base + 1] = f32_to_bf16(xr * sn + xi * c);
}

// ---------------------------------------------------------------------------
// V transpose: vt[kvh*HDIM + d][s] = v[s][kvh*HDIM + d]
// ---------------------------------------------------------------------------
__global__ void transpose_v_kernel(const unsigned short* __restrict__ v,
                                   unsigned short* __restrict__ vt, int n) {
  int idx = blockIdx.x * blockDim.x + threadIdx.x;
  if (idx >= n) return;
  int s = idx % S_LEN;
  int t = idx / S_LEN;                     // kvh*HDIM + d
  vt[idx] = v[(size_t)s * KVDIM + t];
}

// ---------------------------------------------------------------------------
// Flash attention, one wave per 16 query rows, 32-key tiles.
// Q:(S,QDIM) bf16, K:(S,KVDIM) bf16, Vt:(KVDIM,S) bf16, O:(S,QDIM) bf16
// ---------------------------------------------------------------------------
__global__ void __launch_bounds__(256)
attn_wmma(const unsigned short* __restrict__ Q,
          const unsigned short* __restrict__ Kb,
          const unsigned short* __restrict__ Vt,
          unsigned short* __restrict__ Ob) {
  __shared__ __align__(16) unsigned short ldsP[8][16 * 32];  // per-wave P tile

  const int lane = threadIdx.x & 31;
  const int wave = threadIdx.x >> 5;
  const int lm = lane & 15;
  const int lh = lane >> 4;
  const int h = blockIdx.y;
  const int kvh = h / REPEATS;
  const int qbase = blockIdx.x * 128 + wave * 16;

  // Q fragments covering head dim 128 (4 x 16x32)
  v16bf qf[4];
  {
    const unsigned short* qp = Q + (size_t)(qbase + lm) * QDIM + h * HDIM;
    #pragma unroll
    for (int t = 0; t < 4; ++t) qf[t] = load_a_frag(qp + t * 32, lh);
  }

  v8f o[8];
  #pragma unroll
  for (int t = 0; t < 8; ++t) o[t] = v8f_zero();
  float mrow[8], lrow[8];
  #pragma unroll
  for (int r = 0; r < 8; ++r) { mrow[r] = -3.402823466e38f; lrow[r] = 0.f; }

  const int jend = qbase + 15;             // last key this wave can see (causal)
  for (int jb = 0; jb <= jend; jb += 32) {
    // prefetch next key/value tiles into cache (global_prefetch_b8)
    if (jb + 32 <= jend) {
      __builtin_prefetch(Kb + (size_t)(jb + 32 + lane) * KVDIM + kvh * HDIM, 0, 3);
      __builtin_prefetch(Vt + ((size_t)kvh * HDIM + lane * 4) * S_LEN + jb + 32, 0, 3);
    }

    // ---- scores S = Q * K^T  (16 x 32), two 16x16 tiles ----
    v8f s0 = v8f_zero(), s1 = v8f_zero();
    #pragma unroll
    for (int t = 0; t < 4; ++t) {
      const unsigned short* kp0 =
          Kb + (size_t)(jb + lm) * KVDIM + kvh * HDIM + t * 32 + lh * 16;
      const unsigned short* kp1 =
          Kb + (size_t)(jb + 16 + lm) * KVDIM + kvh * HDIM + t * 32 + lh * 16;
      v16bf b0 = load_b_frag(kp0);
      v16bf b1 = load_b_frag(kp1);
      s0 = wmma_bf16(qf[t], b0, s0);
      s1 = wmma_bf16(qf[t], b1, s1);
    }

    // ---- scale + mask + online softmax (elem r -> row r+8*lh, col lane%16;
    //      reduce across the 16-lane half) ----
    float alpha[8];
    #pragma unroll
    for (int r = 0; r < 8; ++r) {
      const int irow = qbase + r + 8 * lh;
      const int j0 = jb + lm;
      const int j1 = jb + 16 + lm;
      float sc0 = s0[r] * ATT_SCALE +
                  ((j0 <= irow && (irow - j0) < WIN) ? 0.f : MASK_NEG);
      float sc1 = s1[r] * ATT_SCALE +
                  ((j1 <= irow && (irow - j1) < WIN) ? 0.f : MASK_NEG);
      float mx = fmaxf(sc0, sc1);
      mx = fmaxf(mx, __shfl_xor(mx, 1, 32));
      mx = fmaxf(mx, __shfl_xor(mx, 2, 32));
      mx = fmaxf(mx, __shfl_xor(mx, 4, 32));
      mx = fmaxf(mx, __shfl_xor(mx, 8, 32));
      float mnew = fmaxf(mrow[r], mx);
      float a = __expf(mrow[r] - mnew);
      float p0 = __expf(sc0 - mnew);
      float p1 = __expf(sc1 - mnew);
      mrow[r] = mnew;
      alpha[r] = a;
      float ps = p0 + p1;
      ps += __shfl_xor(ps, 1, 32);
      ps += __shfl_xor(ps, 2, 32);
      ps += __shfl_xor(ps, 4, 32);
      ps += __shfl_xor(ps, 8, 32);
      lrow[r] = lrow[r] * a + ps;
      const int rloc = r + 8 * lh;
      ldsP[wave][rloc * 32 + lm]      = f32_to_bf16(p0);
      ldsP[wave][rloc * 32 + 16 + lm] = f32_to_bf16(p1);
    }

    // wave-private LDS tile: drain DScnt before the fragment reload
    asm volatile("s_wait_dscnt 0" ::: "memory");

    // reload P as an A-fragment (16x32)
    v16bf pf = load_a_frag(&ldsP[wave][lm * 32], lh);

    // ---- O = diag(alpha)*O + P * V  (8 dim-tiles of 16) ----
    #pragma unroll
    for (int t = 0; t < 8; ++t) {
      const unsigned short* vp =
          Vt + ((size_t)kvh * HDIM + t * 16 + lm) * S_LEN + jb + lh * 16;
      v16bf vf = load_b_frag(vp);
      #pragma unroll
      for (int r = 0; r < 8; ++r) o[t][r] *= alpha[r];
      o[t] = wmma_bf16(pf, vf, o[t]);
    }
  }

  // ---- epilogue: normalize by row sums, store bf16 ----
  #pragma unroll
  for (int t = 0; t < 8; ++t) {
    #pragma unroll
    for (int r = 0; r < 8; ++r) {
      size_t row = (size_t)(qbase + r + 8 * lh);
      Ob[row * QDIM + h * HDIM + t * 16 + lm] = f32_to_bf16(o[t][r] / lrow[r]);
    }
  }
}

// ---------------------------------------------------------------------------
// Launcher
// ---------------------------------------------------------------------------
extern "C" void kernel_launch(void* const* d_in, const int* in_sizes, int n_in,
                              void* d_out, int out_size, void* d_ws, size_t ws_size,
                              hipStream_t stream) {
  const float* x     = (const float*)d_in[0];  // (S, D)
  const float* f_cos = (const float*)d_in[1];  // (S, 64)
  const float* f_sin = (const float*)d_in[2];  // (S, 64)
  const float* wq    = (const float*)d_in[3];  // (QDIM, D)
  const float* wk    = (const float*)d_in[4];  // (KVDIM, D)
  const float* wv    = (const float*)d_in[5];  // (KVDIM, D)
  const float* wo    = (const float*)d_in[6];  // (D, QDIM)
  (void)in_sizes; (void)n_in; (void)out_size; (void)ws_size;

  char* ws = (char*)d_ws;
  size_t off = 0;
  unsigned short* xh   = (unsigned short*)(ws + off); off += (size_t)S_LEN * D_MODEL * 2;
  unsigned short* wqh  = (unsigned short*)(ws + off); off += (size_t)QDIM * D_MODEL * 2;
  unsigned short* wkh  = (unsigned short*)(ws + off); off += (size_t)KVDIM * D_MODEL * 2;
  unsigned short* wvh  = (unsigned short*)(ws + off); off += (size_t)KVDIM * D_MODEL * 2;
  unsigned short* woh  = (unsigned short*)(ws + off); off += (size_t)D_MODEL * QDIM * 2;
  unsigned short* qbuf = (unsigned short*)(ws + off); off += (size_t)S_LEN * QDIM * 2;
  unsigned short* kbuf = (unsigned short*)(ws + off); off += (size_t)S_LEN * KVDIM * 2;
  unsigned short* vbuf = (unsigned short*)(ws + off); off += (size_t)S_LEN * KVDIM * 2;
  unsigned short* vtb  = (unsigned short*)(ws + off); off += (size_t)S_LEN * KVDIM * 2;
  unsigned short* attn = (unsigned short*)(ws + off); off += (size_t)S_LEN * QDIM * 2;

  auto cast = [&](const float* src, unsigned short* dst, int n) {
    cast_f32_bf16_kernel<<<(n + 255) / 256, 256, 0, stream>>>(src, dst, n);
  };
  cast(x,  xh,  S_LEN * D_MODEL);
  cast(wq, wqh, QDIM * D_MODEL);
  cast(wk, wkh, KVDIM * D_MODEL);
  cast(wv, wvh, KVDIM * D_MODEL);
  cast(wo, woh, D_MODEL * QDIM);

  // projections: q = xh * wq^T, k = xh * wk^T, v = xh * wv^T
  gemm_bt_wmma<unsigned short><<<dim3(QDIM / 256, S_LEN / 128), 256, 0, stream>>>(
      xh, wqh, qbuf, S_LEN, QDIM, D_MODEL);
  gemm_bt_wmma<unsigned short><<<dim3(KVDIM / 256, S_LEN / 128), 256, 0, stream>>>(
      xh, wkh, kbuf, S_LEN, KVDIM, D_MODEL);
  gemm_bt_wmma<unsigned short><<<dim3(KVDIM / 256, S_LEN / 128), 256, 0, stream>>>(
      xh, wvh, vbuf, S_LEN, KVDIM, D_MODEL);

  // RoPE on q and k
  {
    int nq = S_LEN * NH * (HDIM / 2);
    int nk = S_LEN * NKVH * (HDIM / 2);
    rope_kernel<<<(nq + 255) / 256, 256, 0, stream>>>(qbuf, f_cos, f_sin, NH, nq);
    rope_kernel<<<(nk + 255) / 256, 256, 0, stream>>>(kbuf, f_cos, f_sin, NKVH, nk);
  }

  // V transpose for contiguous P*V B-fragments
  {
    int n = S_LEN * KVDIM;
    transpose_v_kernel<<<(n + 255) / 256, 256, 0, stream>>>(vbuf, vtb, n);
  }

  // flash attention
  attn_wmma<<<dim3(S_LEN / 128, NH), 256, 0, stream>>>(qbuf, kbuf, vtb, attn);

  // output projection -> f32 d_out
  gemm_bt_wmma<float><<<dim3(D_MODEL / 256, S_LEN / 128), 256, 0, stream>>>(
      attn, woh, (float*)d_out, S_LEN, D_MODEL, QDIM);
}
